// Quantization_Head_16690242912618
// MI455X (gfx1250) — compile-verified, matrix operand order checked by reading, not run on hardware
//
#include <hip/hip_runtime.h>
#include <hip/hip_bf16.h>

typedef __attribute__((ext_vector_type(16))) _Float16 v16h;
typedef __attribute__((ext_vector_type(8)))  float    v8f;
typedef unsigned int u32x4 __attribute__((ext_vector_type(4)));
typedef int          i32x4 __attribute__((ext_vector_type(4)));
typedef int          i32x8 __attribute__((ext_vector_type(8)));

#define N_WORDS 256
#define N_BOOKS 8
#define L_WORD  64
#define FDIM    (N_BOOKS * L_WORD)   /* 512 */
#define TAU_SQ  5.0f
#define BATCH   32768
#define WAVES   4
#define TPB     (WAVES * 32)

#if defined(__has_builtin)
#if __has_builtin(__builtin_amdgcn_tensor_load_to_lds)
#define HAVE_TDM 1
#endif
#endif
#ifndef HAVE_TDM
#define HAVE_TDM 0
#endif

// ISA 7.12.2: 16-bit A-matrix 16x32 — lane group (lane>=16) selects K+8 / K+24,
// VGPRs 0..3 hold K 0..7, VGPRs 4..7 hold K 16..23 (2 halves per VGPR).
__device__ __forceinline__ int frag_k(int i, int lane_hi) {
  int v = i >> 1, h = i & 1;
  return ((v & 3) << 1) + h + (lane_hi << 3) + ((v >> 2) << 4);
}

__device__ __forceinline__ v8f wmma_f16(v16h a, v16h b, v8f c) {
  return __builtin_amdgcn_wmma_f32_16x16x32_f16(false, a, false, b, (short)0, c,
                                                false, false);
}

__device__ __forceinline__ void tdm_wait0() {
#if defined(__has_builtin) && __has_builtin(__builtin_amdgcn_s_wait_tensorcnt)
  __builtin_amdgcn_s_wait_tensorcnt(0);
#else
  asm volatile("s_wait_tensorcnt 0" ::: "memory");
#endif
}

#if HAVE_TDM
// Tensor Data Mover: 2-D f32 tile (tile_d0 x tile_d1) from global (row stride
// stride0 elements) into contiguous LDS at lds_off. D# per ISA 08_async_tensor §8.
__device__ __forceinline__ void tdm_load_2d_f32(unsigned lds_off, const void* gptr,
                                                unsigned tile_d0, unsigned tile_d1,
                                                unsigned long long stride0) {
  unsigned long long ga = (unsigned long long)(uintptr_t)gptr;
  u32x4 g0;
  g0[0] = 1u;                                         // count=1, user D#
  g0[1] = lds_off;                                    // lds_addr (bytes)
  g0[2] = (unsigned)(ga & 0xffffffffu);               // global_addr[31:0]
  g0[3] = (unsigned)((ga >> 32) & 0x01ffffffu) | (2u << 30);  // [56:32] | type=2
  i32x8 g1;
  g1[0] = 0x00020000;                                 // data_size=2 (4B), wg_mask=0
  g1[1] = (int)((tile_d0 & 0xffffu) << 16);           // tensor_dim0[15:0]
  g1[2] = (int)(((tile_d0 >> 16) & 0xffffu) | ((tile_d1 & 0xffffu) << 16));
  g1[3] = (int)(((tile_d1 >> 16) & 0xffffu) | ((tile_d0 & 0xffffu) << 16)); // tile_dim0
  g1[4] = (int)(tile_d1 & 0xffffu);                   // tile_dim1, tile_dim2=0
  g1[5] = (int)(unsigned)(stride0 & 0xffffffffull);   // dim0_stride[31:0]
  g1[6] = (int)(unsigned)((stride0 >> 32) & 0xffffull);
  g1[7] = 0;
  i32x4 z4 = {0, 0, 0, 0};
#if __clang_major__ >= 23
  i32x8 z8 = {0, 0, 0, 0, 0, 0, 0, 0};
  __builtin_amdgcn_tensor_load_to_lds(g0, g1, z4, z4, z8, 0);
#else
  __builtin_amdgcn_tensor_load_to_lds(g0, g1, z4, z4, 0);
#endif
}
#endif

// ---------------------------------------------------------------------------
// Soft quantization: per-book  d = |x|^2 - 2 x.c + |c|^2 ; p = softmax(-tau*d);
// z = p @ C.  One block = one book x 64 batch rows (4 waves x 16-row tiles).
// ---------------------------------------------------------------------------
__global__ void __launch_bounds__(TPB)
softquant_kernel(const float* __restrict__ f,
                 const float* __restrict__ codebooks,
                 float* __restrict__ out_z,
                 float* __restrict__ out_p)
{
  extern __shared__ __align__(16) char smem_raw[];
  float* cbf  = (float*)smem_raw;                  // 256*64 f32 (64 KB)
  float* cc   = cbf + N_WORDS * L_WORD;            // 256
  float* xx   = cc + N_WORDS;                      // WAVES*16
  float* rsinv= xx + WAVES * 16;                   // WAVES*16
  float* ebuf = rsinv + WAVES * 16;                // WAVES*16*256 (64 KB)

  const int book = blockIdx.y;
  const int tid  = threadIdx.x;
  const int wave = tid >> 5;
  const int lane = tid & 31;
  const int nrow = lane & 15;        // row for A, column n for B/C/D
  const int lhi  = lane >> 4;
  const int b0   = blockIdx.x * (WAVES * 16) + wave * 16;

  __builtin_prefetch(f + (size_t)b0 * FDIM + book * L_WORD, 0, 0);

  // codebook slice for this book -> f32 LDS (TDM async DMA; wave 0 issues)
#if HAVE_TDM
  if (wave == 0) {
    tdm_load_2d_f32((unsigned)(uintptr_t)cbf, codebooks + (size_t)book * L_WORD,
                    L_WORD, N_WORDS, FDIM);
    tdm_wait0();
  }
#else
  for (int idx = tid; idx < N_WORDS * L_WORD; idx += TPB) {
    int w = idx >> 6, l = idx & 63;
    cbf[idx] = codebooks[(size_t)w * FDIM + book * L_WORD + l];
  }
#endif
  __syncthreads();

  // |c|^2 from the LDS copy
  for (int w = tid; w < N_WORDS; w += TPB) {
    const float* cw = cbf + w * L_WORD;
    float s = 0.f;
    for (int l = 0; l < L_WORD; ++l) { float v = cw[l]; s += v * v; }
    cc[w] = s;
  }
  __syncthreads();

  // per-wave |x|^2: lane pair (nrow, lhi) each sums half a row, shfl-combine
  {
    const float* xr = f + (size_t)(b0 + nrow) * FDIM + book * L_WORD + lhi * 32;
    float s = 0.f;
    for (int l = 0; l < 32; ++l) { float v = xr[l]; s += v * v; }
    s += __shfl_xor(s, 16);
    if (lhi == 0) xx[wave * 16 + nrow] = s;
  }

  // A fragments: x tile (16 x 64), K split in two 32-chunks
  v16h a0, a1;
  {
    const float* xr = f + (size_t)(b0 + nrow) * FDIM + book * L_WORD;
    for (int i = 0; i < 16; ++i) {
      int k = frag_k(i, lhi);
      a0[i] = (_Float16)xr[k];
      a1[i] = (_Float16)xr[k + 32];
    }
  }

  float* ew = ebuf + wave * 16 * N_WORDS;

  // logits for all 256 words of this book (16 WMMA tiles x 2 k-chunks)
  for (int w0 = 0; w0 < N_WORDS; w0 += 16) {
    v16h bf0, bf1;
    const float* cbn = cbf + (size_t)(w0 + nrow) * L_WORD; // column n = word
    for (int i = 0; i < 16; ++i) {
      int k = frag_k(i, lhi);
      bf0[i] = (_Float16)cbn[k];
      bf1[i] = (_Float16)cbn[k + 32];
    }
    v8f acc = {};
    acc = wmma_f16(a0, bf0, acc);
    acc = wmma_f16(a1, bf1, acc);
    for (int r = 0; r < 8; ++r) {
      int m = r + lhi * 8;
      float d = xx[wave * 16 + m] - 2.0f * acc[r] + cc[w0 + nrow];
      ew[m * N_WORDS + w0 + nrow] = -TAU_SQ * d;
    }
  }
  asm volatile("s_wait_dscnt 0" ::: "memory");

  // stable softmax: each lane scans half a row (128), pairwise shfl-combine
  {
    float* er = ew + nrow * N_WORDS + lhi * 128;
    float mx = -3.4e38f;
    for (int w = 0; w < 128; ++w) mx = fmaxf(mx, er[w]);
    mx = fmaxf(mx, __shfl_xor(mx, 16));
    float s = 0.f;
    for (int w = 0; w < 128; ++w) {
      float e = __expf(er[w] - mx);
      er[w] = e;
      s += e;
    }
    s += __shfl_xor(s, 16);
    if (lhi == 0) rsinv[wave * 16 + nrow] = 1.0f / s;
  }
  asm volatile("s_wait_dscnt 0" ::: "memory");

  // write p (B, K*W): b*2048 + book*256 + w ; b128 stores, 8 words per lane
  for (int m = 0; m < 16; ++m) {
    float inv = rsinv[wave * 16 + m];
    const float4* src = (const float4*)(ew + m * N_WORDS) + lane * 2;
    float4* dst = (float4*)(out_p + (size_t)(b0 + m) * (N_BOOKS * N_WORDS)
                            + book * N_WORDS) + lane * 2;
    float4 v0 = src[0], v1 = src[1];
    v0.x *= inv; v0.y *= inv; v0.z *= inv; v0.w *= inv;
    v1.x *= inv; v1.y *= inv; v1.z *= inv; v1.w *= inv;
    dst[0] = v0; dst[1] = v1;
  }

  // z = (exps @ C) * (1/sum): 4 N-tiles of 16 dims, K=256 in 8 chunks of 32
  for (int l0 = 0; l0 < L_WORD; l0 += 16) {
    v8f acc = {};
    for (int c0 = 0; c0 < N_WORDS; c0 += 32) {
      v16h af, bf;
      for (int i = 0; i < 16; ++i) {
        int k = frag_k(i, lhi);
        af[i] = (_Float16)ew[nrow * N_WORDS + c0 + k];          // A: (m, w)
        bf[i] = (_Float16)cbf[(size_t)(c0 + k) * L_WORD + l0 + nrow]; // B: (w, l)
      }
      acc = wmma_f16(af, bf, acc);
    }
    for (int r = 0; r < 8; ++r) {
      int m = r + lhi * 8;
      out_z[(size_t)(b0 + m) * FDIM + book * L_WORD + l0 + nrow] =
          acc[r] * rsinv[wave * 16 + m];
    }
  }
}

// ---------------------------------------------------------------------------
// MLP: h = relu(f_split @ W1 + b1); out = log_softmax(h @ W2 + b2)
// f_split rows = 262144 x 64 (contiguous reshape of f). 64 rows per block.
// ---------------------------------------------------------------------------
__global__ void __launch_bounds__(TPB)
mlp_kernel(const float* __restrict__ f,
           const float* __restrict__ W1, const float* __restrict__ b1,
           const float* __restrict__ W2, const float* __restrict__ b2,
           float* __restrict__ out_lg)
{
  extern __shared__ __align__(16) char smem_raw[];
  float*    w1f = (float*)smem_raw;               // 64*256 f32 (64 KB)
  _Float16* w2s = (_Float16*)(w1f + 64 * 256);    // 256*16 f16 padded (8 KB)
  _Float16* hb  = w2s + 256 * 16;                 // WAVES*16*256 f16 (32 KB)
  float* b1s = (float*)(hb + WAVES * 16 * 256);   // 256
  float* b2s = b1s + 256;                         // 16
  float* lg  = b2s + 16;                          // WAVES*16*16

  const int tid  = threadIdx.x;
  const int wave = tid >> 5;
  const int lane = tid & 31;
  const int nrow = lane & 15;
  const int lhi  = lane >> 4;
  const size_t r0 = (size_t)blockIdx.x * (WAVES * 16) + wave * 16;

  __builtin_prefetch(f + r0 * L_WORD, 0, 0);

  // W1 (64x256 f32, contiguous) -> LDS via TDM as a 1-D tile
#if HAVE_TDM
  if (wave == 0) {
    tdm_load_2d_f32((unsigned)(uintptr_t)w1f, W1, 64 * 256, 1, 64 * 256);
    tdm_wait0();
  }
#else
  for (int idx = tid; idx < 64 * 256; idx += TPB) w1f[idx] = W1[idx];
#endif
  for (int idx = tid; idx < 256 * 16; idx += TPB) {
    int kk = idx >> 4, n = idx & 15;
    w2s[idx] = (n < N_BOOKS) ? (_Float16)W2[kk * N_BOOKS + n] : (_Float16)0.f;
  }
  for (int idx = tid; idx < 256; idx += TPB) b1s[idx] = b1[idx];
  if (tid < 16) b2s[tid] = (tid < N_BOOKS) ? b2[tid] : 0.f;
  __syncthreads();

  // A fragments for the 16 x 64 input tile
  v16h a0, a1;
  {
    const float* xr = f + (r0 + nrow) * L_WORD;
    for (int i = 0; i < 16; ++i) {
      int k = frag_k(i, lhi);
      a0[i] = (_Float16)xr[k];
      a1[i] = (_Float16)xr[k + 32];
    }
  }

  _Float16* hw = hb + wave * 16 * 256;

  // layer 1: 16 N-tiles x (K=64 -> 2 WMMAs), relu -> f16 LDS
  for (int n0 = 0; n0 < 256; n0 += 16) {
    v16h bf0, bf1;
    for (int i = 0; i < 16; ++i) {
      int k = frag_k(i, lhi);
      bf0[i] = (_Float16)w1f[k * 256 + n0 + nrow];
      bf1[i] = (_Float16)w1f[(k + 32) * 256 + n0 + nrow];
    }
    v8f acc = {};
    acc = wmma_f16(a0, bf0, acc);
    acc = wmma_f16(a1, bf1, acc);
    for (int r = 0; r < 8; ++r) {
      int m = r + lhi * 8;
      float h = acc[r] + b1s[n0 + nrow];
      hw[m * 256 + n0 + nrow] = (_Float16)fmaxf(h, 0.f);
    }
  }
  asm volatile("s_wait_dscnt 0" ::: "memory");

  // layer 2: K=256 in 8 chunks, N padded 8->16 (cols 8..15 ignored)
  v8f acc2 = {};
  for (int c0 = 0; c0 < 256; c0 += 32) {
    v16h af, bf;
    for (int i = 0; i < 16; ++i) {
      int k = frag_k(i, lhi);
      af[i] = hw[nrow * 256 + c0 + k];
      bf[i] = w2s[(c0 + k) * 16 + nrow];
    }
    acc2 = wmma_f16(af, bf, acc2);
  }
  float* lw = lg + wave * 16 * 16;
  for (int r = 0; r < 8; ++r) {
    int m = r + lhi * 8;
    lw[m * 16 + nrow] = acc2[r] + b2s[nrow];
  }
  asm volatile("s_wait_dscnt 0" ::: "memory");

  // per-row log-softmax over 8 logits
  if (lane < 16) {
    float* lr = lw + lane * 16;
    float mx = -3.4e38f;
    for (int j = 0; j < N_BOOKS; ++j) mx = fmaxf(mx, lr[j]);
    float s = 0.f;
    for (int j = 0; j < N_BOOKS; ++j) s += __expf(lr[j] - mx);
    float lse = mx + __logf(s);
    float* op = out_lg + (r0 + lane) * N_BOOKS;
    for (int j = 0; j < N_BOOKS; ++j) op[j] = lr[j] - lse;
  }
}

extern "C" void kernel_launch(void* const* d_in, const int* in_sizes, int n_in,
                              void* d_out, int out_size, void* d_ws, size_t ws_size,
                              hipStream_t stream) {
  (void)in_sizes; (void)n_in; (void)out_size; (void)d_ws; (void)ws_size;
  const float* f  = (const float*)d_in[0];
  const float* cb = (const float*)d_in[1];
  const float* W1 = (const float*)d_in[2];
  const float* b1 = (const float*)d_in[3];
  const float* W2 = (const float*)d_in[4];
  const float* b2 = (const float*)d_in[5];

  float* out = (float*)d_out;
  const size_t nf = (size_t)BATCH * FDIM;                 // 16,777,216
  float* out_f = out;
  float* out_z = out + nf;
  float* out_p = out + 2 * nf;
  float* out_l = out + 2 * nf + (size_t)BATCH * N_BOOKS * N_WORDS;

  // output 0: f passthrough
  hipMemcpyAsync(out_f, f, nf * sizeof(float), hipMemcpyDeviceToDevice, stream);

  // soft quantization: z and p
  {
    dim3 grid(BATCH / (WAVES * 16), N_BOOKS);
    size_t smem = (size_t)N_WORDS * L_WORD * sizeof(float)      // cbf
                + N_WORDS * sizeof(float)                       // cc
                + 2 * WAVES * 16 * sizeof(float)                // xx, rsinv
                + (size_t)WAVES * 16 * N_WORDS * sizeof(float); // ebuf
    softquant_kernel<<<grid, TPB, smem, stream>>>(f, cb, out_z, out_p);
  }

  // MLP classifier
  {
    dim3 grid((BATCH * N_BOOKS) / (WAVES * 16));
    size_t smem = (size_t)64 * 256 * sizeof(float)               // w1f
                + (size_t)256 * 16 * sizeof(_Float16)            // w2s
                + (size_t)WAVES * 16 * 256 * sizeof(_Float16)    // hb
                + (256 + 16) * sizeof(float)                     // b1s, b2s
                + (size_t)WAVES * 16 * 16 * sizeof(float);       // lg
    mlp_kernel<<<grid, TPB, smem, stream>>>(f, W1, b1, W2, b2, out_l);
  }
}
